// NearFieldHead_10170482557399
// MI455X (gfx1250) — matrix operand order repaired and checked by reading, not compiled, not run on hardware
//
#include <hip/hip_runtime.h>
#include <hip/hip_bf16.h>
#include <math.h>

// ---------------------------------------------------------------------------
// NearFieldHead fused attention for gfx1250 (MI455X, wave32, WMMA).
// One wave per batch element. Softmax-invariance removes the H=64 dimension:
//   scores ~ Hq (Wq2 Wk2^T) Hk^T + (Hk . (Wk2 qb2))[l]   (per-row consts drop)
// 6 WMMAs per batch, chained so D-layout == next A-layout.  The final WMMA
// computes S^T (lane = r, regs = l) so softmax-over-l is register-local; the
// Hk.v term is folded into the B operand and the distance bias enters as a
// rank-1 WMMA (compensated f16 pair) through the C accumulator.
// ---------------------------------------------------------------------------

typedef __attribute__((ext_vector_type(16))) _Float16 v16h;
typedef __attribute__((ext_vector_type(8)))  float    v8f;

#define WPB     8      // waves per block (256 threads)
#define BLOCK   256
#define NBLOCKS 1024   // 8192 waves, grid-stride over 65536 batches

union H2U    { unsigned u; _Float16 h[2]; };
union F16x16 { v16h v; unsigned u[8]; };

// A-operand element->K mapping (16-bit A 16x32, ISA 7.12.2):
// lanes 0-15 hold K in {0..7,16..23}, lanes 16-31 hold K in {8..15,24..31}
static __device__ __forceinline__ int a_kidx(int e, int half) {
  return e + 8 * half + (e >= 8 ? 8 : 0);
}

static __device__ __forceinline__ v8f wmma16(v16h a, v16h b, v8f c) {
  return __builtin_amdgcn_wmma_f32_16x16x32_f16(false, a, false, b,
                                                (short)0, c, false, false);
}

__global__ __launch_bounds__(BLOCK) void nearfield_attn_kernel(
    const float* __restrict__ rss_t,      // [B,12]
    const float* __restrict__ led_f,      // [B,16,8]
    const float* __restrict__ led_p,      // [B,16,3]
    const float* __restrict__ prev_pos,   // [B,3]
    const int*   __restrict__ freq_mask,  // [12,16]
    const float* __restrict__ gw1, const float* __restrict__ gb1,
    const float* __restrict__ gw2, const float* __restrict__ gb2,
    const float* __restrict__ qw1, const float* __restrict__ qb1,
    const float* __restrict__ qw2, const float* __restrict__ qb2,
    const float* __restrict__ kw1, const float* __restrict__ kb1,
    const float* __restrict__ kw2,
    const float* __restrict__ sigma_p,
    float* __restrict__ out_attn,         // [B,12,16]
    float* __restrict__ out_iw,           // [B,12]
    int nBatch)
{
  // ---- block init: W = qw2 * kw2^T [32x32], vvec = kw2 * qb2 [32] ---------
  __shared__ float s_W[32 * 32];
  __shared__ float s_vvec[32];

  const int tid = threadIdx.x;
  for (int idx = tid; idx < 32 * 32; idx += BLOCK) {
    const int i = idx >> 5, j = idx & 31;
    const float* qi = qw2 + i * 64;
    const float* kj = kw2 + j * 64;
    float acc = 0.0f;
#pragma unroll 8
    for (int h = 0; h < 64; ++h) acc = fmaf(qi[h], kj[h], acc);
    s_W[idx] = acc;                       // s_W[i*32 + j]
  }
  for (int j = tid; j < 32; j += BLOCK) {
    const float* kj = kw2 + j * 64;
    float acc = 0.0f;
#pragma unroll 8
    for (int h = 0; h < 64; ++h) acc = fmaf(kj[h], qb2[h], acc);
    s_vvec[j] = acc;
  }
  __syncthreads();

  const int lane = tid & 31;
  const int half = lane >> 4;             // 16-lane half
  const int m    = lane & 15;             // r for A-side cols / l for B-side
  const int w    = tid >> 5;

  // ---- loop-invariant operand fragments & hoisted per-lane constants -----
  v16h Akw1[2];   // A' = kw1^T chunk t: rows khid=16t+m, K = feature (11 pad 32)
  v16h AWt[2];    // A  = W^T  chunk t: rows j   =16t+m, K = i (Hq hidden)
#pragma unroll
  for (int t = 0; t < 2; ++t) {
#pragma unroll
    for (int e = 0; e < 16; ++e) {
      const int f = e + 16 * half;
      Akw1[t][e] = (f < 11) ? (_Float16)kw1[f * 32 + 16 * t + m]
                            : (_Float16)0.0f;
      AWt[t][e]  = (_Float16)s_W[f * 32 + 16 * t + m];  // W[i][j], i=f, j=16t+m
    }
  }
  // B_ones: ones in K rows 0,1 (rank-1 bias broadcast), loop-invariant
  F16x16 FBones;
#pragma unroll
  for (int u = 0; u < 8; ++u) FBones.u[u] = 0u;
  {
    H2U one2; one2.h[0] = (_Float16)1.0f; one2.h[1] = (_Float16)1.0f;
    FBones.u[0] = half ? 0u : one2.u;     // k=0,1 live in half-0 lanes
  }
  // A_dist: zero except K slots 0,1 (set per batch), persistent zeros
  F16x16 FAd;
#pragma unroll
  for (int u = 0; u < 8; ++u) FAd.u[u] = 0u;

  float qw1B[16], qb1B[16];               // Hq^T B-view: i = 16*half + e
#pragma unroll
  for (int e = 0; e < 16; ++e) {
    qw1B[e] = half ? qw1[16 + e] : qw1[e];
    qb1B[e] = half ? qb1[16 + e] : qb1[e];
  }
  float kb1a[8], kb1b[8], vva[8], vvb[8]; // index = v + 8*half (+16 for 'b')
#pragma unroll
  for (int v = 0; v < 8; ++v) {
    kb1a[v] = half ? kb1[v + 8]  : kb1[v];
    kb1b[v] = half ? kb1[v + 24] : kb1[v + 16];
    vva[v]  = half ? s_vvec[v + 8]  : s_vvec[v];
    vvb[v]  = half ? s_vvec[v + 24] : s_vvec[v + 16];
  }
  unsigned maskbits = 0;                  // bit v: row r = m, col l = v+8*half
#pragma unroll
  for (int v = 0; v < 8; ++v) {
    const int mk = (m < 12) ? freq_mask[m * 16 + v + 8 * half] : 1;
    if (mk) maskbits |= (1u << v);
  }
  const float sg = sigma_p[0];
  const float inv2s2 = 1.0f / (2.0f * sg * sg);
  const float negInf = -3.402823466e38f;  // finfo(f32).min
  const v8f zero = {};

  const int waveGid = blockIdx.x * WPB + w;
  const int nWaves  = gridDim.x * WPB;

  for (int b = waveGid; b < nBatch; b += nWaves) {
    if (b + nWaves < nBatch)              // stream hint for the big input
      __builtin_prefetch(led_f + (size_t)(b + nWaves) * 128, 0, 1);

    // ---- gate MLP + hard threshold (lane -> frequency r = m) -------------
    const int r_ld = (m < 12) ? m : 11;
    const float rss = rss_t[(size_t)b * 12 + r_ld];
    float g = gb2[0];
#pragma unroll
    for (int j = 0; j < 16; ++j)
      g = fmaf(fmaxf(fmaf(rss, gw1[j], gb1[j]), 0.0f), gw2[j], g);
    const float gate = 1.0f / (1.0f + __expf(-g));
    const float iwv  = (rss > 0.5f) ? gate : 0.0f;
    const float x    = rss * iwv;
    if (half == 0 && m < 12) out_iw[(size_t)b * 12 + m] = iwv;

    // ---- B operand: Hq^T (lane n = r, K = i = 16*half+e), pure VALU ------
    v16h Bhq;
#pragma unroll
    for (int e = 0; e < 16; ++e)
      Bhq[e] = (_Float16)fmaxf(fmaf(x, qw1B[e], qb1B[e]), 0.0f);

    // ---- B operand: led_input^T (lane n = l, K = feature = 16*half+e) ----
    const float* lf = led_f + (size_t)b * 128 + m * 8;
    const float* lp = led_p + (size_t)b * 48 + m * 3;
    v16h Bled;
#pragma unroll
    for (int e = 0; e < 16; ++e) {
      float v = 0.0f;
      if (e < 8)       v = lf[e];
      else if (e < 11) v = lp[e - 8];
      Bled[e] = half ? (_Float16)0.0f : (_Float16)v;  // K rows 16..31 are pad
    }

    // ---- per-l distance bias, compensated f16 pair in A_dist K=0,1 -------
    float dsq = 0.0f;
#pragma unroll
    for (int c = 0; c < 3; ++c) {
      const float d = prev_pos[(size_t)b * 3 + c] - lp[c];
      dsq = fmaf(d, d, dsq);
    }
    const float dbn = -(dsq * inv2s2);    // bias for column l = m (negated)
    {
      H2U dp;
      dp.h[0] = (_Float16)dbn;                         // hi part
      dp.h[1] = (_Float16)(dbn - (float)dp.h[0]);      // lo compensation
      FAd.u[0] = half ? 0u : dp.u;        // rows l, K slots 0,1 (half-0 only)
    }
    // Dbias[l][r] = -dist[l] for every r (rank-1 broadcast WMMA)
    v8f Dbias = wmma16(FAd.v, FBones.v, zero);

    // ---- layer1^T: Hk^T[khid][l] = relu(kw1^T led^T + kb1), 2 WMMAs ------
    v8f d0 = wmma16(Akw1[0], Bled, zero); // D: lane = l, rows khid = v+8h
    v8f d1 = wmma16(Akw1[1], Bled, zero); //    rows khid = 16+v+8h
    float h0[8], h1[8];
#pragma unroll
    for (int v = 0; v < 8; ++v) {
      h0[v] = fmaxf(d0[v] + kb1a[v], 0.0f);
      h1[v] = fmaxf(d1[v] + kb1b[v], 0.0f);
    }
    // D-layout == A-layout: Hk as A operand [16 l x 32 khid], packed build
    F16x16 FAhk;
#pragma unroll
    for (int u = 0; u < 4; ++u) {
      H2U a; a.h[0] = (_Float16)h0[2 * u]; a.h[1] = (_Float16)h0[2 * u + 1];
      H2U c; c.h[0] = (_Float16)h1[2 * u]; c.h[1] = (_Float16)h1[2 * u + 1];
      FAhk.u[u]     = a.u;
      FAhk.u[u + 4] = c.u;
    }

    // ---- T1^T = W^T Hq^T, 2 WMMAs (D: lane = r, rows j = v+8h (+16)) -----
    v8f t0 = wmma16(AWt[0], Bhq, zero);
    v8f t1 = wmma16(AWt[1], Bhq, zero);

    // ---- B operand: (T1^T + vvec.1^T), rows j = 8*half+e (+8 if e>=8) ----------------
    // Fold Hk.vvec into the matmul; packed u32 cross-half exchange.
    unsigned t0p[4], t1p[4];
#pragma unroll
    for (int u = 0; u < 4; ++u) {
      H2U a; a.h[0] = (_Float16)(t0[2 * u]     + vva[2 * u]);
             a.h[1] = (_Float16)(t0[2 * u + 1] + vva[2 * u + 1]);
      H2U c; c.h[0] = (_Float16)(t1[2 * u]     + vvb[2 * u]);
             c.h[1] = (_Float16)(t1[2 * u + 1] + vvb[2 * u + 1]);
      t0p[u] = a.u; t1p[u] = c.u;
    }
    F16x16 FBt;
#pragma unroll
    for (int u = 0; u < 4; ++u) {
      const unsigned t0x = (unsigned)__shfl_xor((int)t0p[u], 16, 32);
      const unsigned t1x = (unsigned)__shfl_xor((int)t1p[u], 16, 32);
      FBt.u[u]     = t0p[u];  // elements 0..7  (K 8h+0..7)
      FBt.u[u + 4] = t1p[u];  // elements 8..15 (K 16+8h+0..7)
    }

    // ---- S^T = Hk (T1^T + v.1^T) + Dbias: lane = r, regs = l = v+8h ------
    v8f S = wmma16(FAhk.v, FBt.v, Dbias);

    // ---- mask + register-local softmax over l ----------------------------
    float sc[8];
#pragma unroll
    for (int v = 0; v < 8; ++v)
      sc[v] = ((maskbits >> v) & 1u) ? S[v] : negInf;

    float mx = sc[0];
#pragma unroll
    for (int v = 1; v < 8; ++v) mx = fmaxf(mx, sc[v]);
    mx = fmaxf(mx, __shfl_xor(mx, 16, 32));

    float p[8];
    float sum = 0.0f;
#pragma unroll
    for (int v = 0; v < 8; ++v) { p[v] = __expf(sc[v] - mx); sum += p[v]; }
    sum += __shfl_xor(sum, 16, 32);
    const float inv = __frcp_rn(sum);

    if (m < 12) {
      float4 o0 = make_float4(p[0] * inv, p[1] * inv, p[2] * inv, p[3] * inv);
      float4 o1 = make_float4(p[4] * inv, p[5] * inv, p[6] * inv, p[7] * inv);
      float* op = out_attn + (size_t)b * 192 + m * 16 + 8 * half;
      *(float4*)op       = o0;            // l = 8*half + 0..3
      *((float4*)op + 1) = o1;            // l = 8*half + 4..7
    }
  }
}

extern "C" void kernel_launch(void* const* d_in, const int* in_sizes, int n_in,
                              void* d_out, int out_size, void* d_ws, size_t ws_size,
                              hipStream_t stream) {
  (void)n_in; (void)d_ws; (void)ws_size; (void)out_size;
  const float* rss_t     = (const float*)d_in[0];
  const float* led_f     = (const float*)d_in[1];
  const float* led_p     = (const float*)d_in[2];
  const float* prev_pos  = (const float*)d_in[3];
  const int*   freq_mask = (const int*)  d_in[4];
  const float* gw1 = (const float*)d_in[5];
  const float* gb1 = (const float*)d_in[6];
  const float* gw2 = (const float*)d_in[7];
  const float* gb2 = (const float*)d_in[8];
  const float* qw1 = (const float*)d_in[9];
  const float* qb1 = (const float*)d_in[10];
  const float* qw2 = (const float*)d_in[11];
  const float* qb2 = (const float*)d_in[12];
  const float* kw1 = (const float*)d_in[13];
  const float* kb1 = (const float*)d_in[14];
  const float* kw2 = (const float*)d_in[15];
  // d_in[16] (kb2) drops out of the softmax algebra entirely.
  const float* sigma = (const float*)d_in[17];

  const int nBatch = in_sizes[0] / 12;          // rss_t is [B,12,1]
  float* out_attn = (float*)d_out;              // [B,12,16]
  float* out_iw   = out_attn + (size_t)nBatch * 12 * 16; // [B,12]

  int blocks = (nBatch + WPB - 1) / WPB;
  if (blocks > NBLOCKS) blocks = NBLOCKS;
  if (blocks < 1) blocks = 1;

  nearfield_attn_kernel<<<blocks, BLOCK, 0, stream>>>(
      rss_t, led_f, led_p, prev_pos, freq_mask,
      gw1, gb1, gw2, gb2, qw1, qb1, qw2, qb2, kw1, kb1, kw2, sigma,
      out_attn, out_iw, nBatch);
}
